// NERRelationModel_11553462026364
// MI455X (gfx1250) — matrix operand order repaired and checked by reading, not compiled, not run on hardware
//
#include <hip/hip_runtime.h>
#include <hip/hip_bf16.h>

// Model dims (from reference)
#define BB 64
#define SS 512
#define HH 768
#define EE 32
#define LL 8
#define PP 128
#define NNER 11
#define FOUNDED_BY 2

typedef __attribute__((ext_vector_type(16))) __bf16 v16bf;
typedef __attribute__((ext_vector_type(8)))  __bf16 v8bf;
typedef __attribute__((ext_vector_type(4)))  __bf16 v4bf;
typedef __attribute__((ext_vector_type(8)))  float  v8f;
typedef __attribute__((ext_vector_type(4)))  unsigned int u32x4;
typedef __attribute__((ext_vector_type(8)))  int i32x8;
typedef __attribute__((ext_vector_type(4)))  int i32x4;

__device__ __forceinline__ v4bf cvt4(float4 f) {
  v4bf o;
  o[0] = (__bf16)f.x; o[1] = (__bf16)f.y; o[2] = (__bf16)f.z; o[3] = (__bf16)f.w;
  return o;
}

__device__ __forceinline__ v16bf cat16(v8bf lo, v8bf hi) {
  return __builtin_shufflevector(lo, hi, 0,1,2,3,4,5,6,7,8,9,10,11,12,13,14,15);
}

// -----------------------------------------------------------------------------
// Kernel 1: ner_logits = relu(seq @ Wn1 + bn1) @ Wn2 + bn2
// 128 threads = 4 waves; tile M=32, N=256, K staged 64/iter as bf16.
// B staged pre-swizzled into fragment order (one 32B LDS load per fragment).
// All 8 B fragments are loaded into registers first, then the 8 WMMAs issue
// back-to-back behind a single s_wait_dscnt.
// -----------------------------------------------------------------------------
__global__ __launch_bounds__(128) void ner_kernel(
    const float* __restrict__ seq, const float* __restrict__ Wn1,
    const float* __restrict__ bn1, const float* __restrict__ Wn2,
    const float* __restrict__ bn2, float* __restrict__ out)
{
  const int tid = threadIdx.x;
  const int m0  = blockIdx.x * 32;

  __shared__ __align__(32) char smem[45440];
  __bf16* Bsw      = reinterpret_cast<__bf16*>(smem);                  // [2kt][16nt][32 lane][16]
  __bf16 (*As)[72] = reinterpret_cast<__bf16(*)[72]>(smem + 32768);    // 32x72 bf16
  float (*Y1)[264]   = reinterpret_cast<float(*)[264]>(smem);          // overlay
  float (*Wn2T)[264] = reinterpret_cast<float(*)[264]>(smem + 33792);  // overlay

  const int wave = tid >> 5, lane = tid & 31;
  const int mt  = wave & 1;
  const int nh  = wave >> 1;
  const int lhi = lane >> 4, llo = lane & 15;

  v8f acc[8];
  for (int a = 0; a < 8; ++a)
    for (int k = 0; k < 8; ++k) acc[a][k] = 0.f;

  for (int k0 = 0; k0 < HH; k0 += 64) {
    // stage A (32 x 64) as bf16: float4 reads, v4bf (b64) LDS writes
    for (int i = tid; i < 32 * 16; i += 128) {
      int r = i >> 4, c4 = i & 15;
      const float* sp = &seq[(size_t)(m0 + r) * HH + k0 + c4 * 4];
      if (k0 + 64 < HH) __builtin_prefetch(sp + 64, 0, 1);
      *(v4bf*)&As[r][c4 * 4] = cvt4(*(const float4*)sp);
    }
    // stage Wn1 (64 x 256) swizzled into fragment order
    for (int g = tid; g < 16 * 256; g += 128) {
      int n = g & 255, k = (g >> 8) * 4;
      const float* src = &Wn1[(size_t)(k0 + k) * 256 + n];
      v4bf v;
      v[0] = (__bf16)src[0];   v[1] = (__bf16)src[256];
      v[2] = (__bf16)src[512]; v[3] = (__bf16)src[768];
      int ktile = k >> 5, lh = (k >> 3) & 1, i0 = (k & 7) + ((k >> 4) & 1) * 8;
      int ntile = n >> 4, ll = n & 15;
      *(v4bf*)&Bsw[(((ktile * 16 + ntile) * 32 + lh * 16 + ll) << 4) + i0] = v;
    }
    __syncthreads();

    for (int kt = 0; kt < 2; ++kt) {
      int kk = kt * 32;
      v8bf alo = *(const v8bf*)&As[mt * 16 + llo][kk + lhi * 8];
      v8bf ahi = *(const v8bf*)&As[mt * 16 + llo][kk + lhi * 8 + 16];
      v16bf af = cat16(alo, ahi);
      const __bf16* bbase = &Bsw[((kt * 16 + nh * 8) * 32 + lane) << 4];
      v16bf bfr[8];
      #pragma unroll
      for (int a = 0; a < 8; ++a)
        bfr[a] = *(const v16bf*)(bbase + a * 512);   // burst of ds_load_b128
      #pragma unroll
      for (int a = 0; a < 8; ++a)                    // back-to-back WMMAs
        acc[a] = __builtin_amdgcn_wmma_f32_16x16x32_bf16(
            false, af, false, bfr[a], (short)0, acc[a], false, false);
    }
    __syncthreads();
  }

  // bias + relu into the LDS overlay; stage Wn2^T alongside
  for (int a = 0; a < 8; ++a) {
    int col = nh * 128 + a * 16 + llo;
    float bias = bn1[col];
    for (int r = 0; r < 8; ++r) {
      int row = mt * 16 + r + 8 * lhi;
      float y = acc[a][r] + bias;
      Y1[row][col] = y > 0.f ? y : 0.f;
    }
  }
  for (int i = tid; i < NNER * 256; i += 128) {
    int c = i >> 8, n = i & 255;
    Wn2T[c][n] = Wn2[n * NNER + c];
  }
  __syncthreads();

  for (int i = tid; i < 32 * NNER; i += 128) {
    int r = i / NNER, c = i % NNER;
    float s = bn2[c];
    const float4* yv = (const float4*)Y1[r];
    const float4* wv = (const float4*)Wn2T[c];
    for (int n4 = 0; n4 < 64; ++n4) {
      float4 y = yv[n4], w = wv[n4];
      s += y.x * w.x + y.y * w.y + y.z * w.z + y.w * w.w;
    }
    out[(size_t)(m0 + r) * NNER + c] = s;
  }
}

// -----------------------------------------------------------------------------
// Kernel 2: span pooling. One block per (b,e).
// The 8x768 f32 token block is CONTIGUOUS in memory -> Tensor Data Mover copies
// it into LDS asynchronously (one TDM op replaces three global re-reads).
// -----------------------------------------------------------------------------
__global__ __launch_bounds__(256) void span_pool_kernel(
    const float* __restrict__ seq, const int* __restrict__ starts,
    const int* __restrict__ lens, const int* __restrict__ types,
    const float* __restrict__ type_emb, float* __restrict__ x)
{
  const int be = blockIdx.x;
  const int b = be >> 5;
  const int tid = threadIdx.x;

  __shared__ __align__(16) float toks[LL * HH];   // 24 KB staged token block
  __shared__ float meansh[HH];
  __shared__ float part[LL][256];
  __shared__ float wsh[LL];

  const int start = starts[be];
  const int slen  = lens[be];            // in [0, L-1]; mask: l <= slen
  const float cnt = (float)(slen + 1);
  const float* base = seq + ((size_t)b * SS + start) * HH;

#if __has_builtin(__builtin_amdgcn_tensor_load_to_lds)
  if (tid < 32) {                         // wave 0 issues the TDM copy
    const unsigned long long ga = (unsigned long long)(const void*)base;
    const unsigned int lds_off = (unsigned int)(unsigned long long)(const void*)&toks[0];
    u32x4 g0;
    g0[0] = 1u;                                   // count=1 valid user descriptor
    g0[1] = lds_off;                              // lds_addr (bytes)
    g0[2] = (unsigned int)ga;                     // global_addr[31:0]
    g0[3] = (unsigned int)(ga >> 32) | (2u << 30);// global_addr[56:32] | type=2
    i32x8 g1;
    g1[0] = 0x20000;                              // data_size=2 -> 4-byte elems
    g1[1] = (int)(6144u << 16);                   // tensor_dim0 = 6144 (lo16)
    g1[2] = (int)(1u << 16);                      // tensor_dim0 hi=0, tensor_dim1=1
    g1[3] = (int)(6144u << 16);                   // tile_dim0 = 6144
    g1[4] = 0;                                    // tile_dim1=0, tile_dim2=0 (1-D)
    g1[5] = 6144;                                 // tensor_dim0_stride lo32
    g1[6] = 0;
    g1[7] = 0;
    i32x4 gz = {0, 0, 0, 0};
#if __has_include(<hip/amd_detail/amd_gfx1250_TDM.h>)
    i32x8 z8 = {0, 0, 0, 0, 0, 0, 0, 0};
    __builtin_amdgcn_tensor_load_to_lds(g0, g1, gz, gz, z8, 0);
#else
    __builtin_amdgcn_tensor_load_to_lds(g0, g1, gz, gz, 0);
#endif
    __builtin_amdgcn_s_wait_tensorcnt(0);
  }
  __syncthreads();
#else
  for (int i = tid; i < LL * HH; i += 256) toks[i] = base[i];
  __syncthreads();
#endif

  for (int h = tid; h < HH; h += 256) {
    float s = 0.f;
    for (int l = 0; l <= slen; ++l) s += toks[l * HH + h];
    meansh[h] = s / cnt;
  }
  __syncthreads();

  float p[LL];
  for (int l = 0; l < LL; ++l) p[l] = 0.f;
  for (int h = tid; h < HH; h += 256) {
    float m = meansh[h];
    for (int l = 0; l < LL; ++l) p[l] += toks[l * HH + h] * m;
  }
  for (int l = 0; l < LL; ++l) part[l][tid] = p[l];
  __syncthreads();

  if (tid == 0) {
    float sc[LL];
    float mx = -1e30f;
    for (int l = 0; l <= slen; ++l) {
      float s = 0.f;
      for (int t = 0; t < 256; ++t) s += part[l][t];
      sc[l] = s;
      mx = fmaxf(mx, s);
    }
    float den = 0.f;
    for (int l = 0; l < LL; ++l) {
      float w = (l <= slen) ? __expf(sc[l] - mx) : 0.f;
      wsh[l] = w;
      den += w;
    }
    float inv = 1.f / den;
    for (int l = 0; l < LL; ++l) wsh[l] *= inv;
  }
  __syncthreads();

  const int ty = types[be];
  for (int h = tid; h < HH; h += 256) {
    float s = 0.f;
    for (int l = 0; l < LL; ++l) s += wsh[l] * toks[l * HH + h];
    x[(size_t)be * HH + h] = s + type_emb[ty * HH + h];
  }
}

// -----------------------------------------------------------------------------
// Kernel 3: per-batch cls contribution  cc[b][n] = sum_f cls[b][f]*Wr1[160+f][n]
// -----------------------------------------------------------------------------
__global__ __launch_bounds__(256) void cls_kernel(
    const float* __restrict__ seq, const float* __restrict__ Wr1,
    float* __restrict__ cc)
{
  const int b = blockIdx.x;
  const int n = threadIdx.x;   // 256
  const float* cls = seq + (size_t)b * SS * HH;   // token 0
  float s = 0.f;
  for (int f = 0; f < HH; ++f) s += cls[f] * Wr1[(160 + f) * 256 + n];
  cc[b * 256 + n] = s;
}

// -----------------------------------------------------------------------------
// Kernel 4: GAT layer 1 (2 heads, concat) + LN + ELU. One block per batch item.
// -----------------------------------------------------------------------------
__global__ __launch_bounds__(256) void gat1_kernel(
    const float* __restrict__ x, const float* __restrict__ W1,
    const float* __restrict__ a_src1, const float* __restrict__ a_dst1,
    const float* __restrict__ b1, const float* __restrict__ g1,
    const float* __restrict__ be1, float* __restrict__ h1out)
{
  const int b = blockIdx.x;
  const int tid = threadIdx.x;

  __shared__ float hsh[2][EE][128];
  __shared__ float ssh[2][EE], sdh[2][EE];
  __shared__ float alph[2][EE][EE];
  __shared__ float rsum[EE][8], rsq[EE][8];

  for (int i = tid; i < 2 * EE * 128; i += 256) {
    int head = i >> 12, e = (i >> 7) & 31, o = i & 127;
    const float* xp = x + ((size_t)b * EE + e) * HH;
    const float* wp = W1 + (size_t)head * HH * 128 + o;
    float s = 0.f;
    for (int f = 0; f < HH; ++f) s += xp[f] * wp[f * 128];
    hsh[head][e][o] = s;
  }
  __syncthreads();

  if (tid < 64) {
    int head = tid >> 5, e = tid & 31;
    float s1 = 0.f, s2 = 0.f;
    for (int o = 0; o < 128; ++o) {
      float hv = hsh[head][e][o];
      s1 += hv * a_src1[head * 128 + o];
      s2 += hv * a_dst1[head * 128 + o];
    }
    ssh[head][e] = s1;
    sdh[head][e] = s2;
  }
  __syncthreads();

  if (tid < 64) {   // softmax over j of leaky_relu(sd[i]+ss[j])
    int head = tid >> 5, i = tid & 31;
    float ev[EE], mx = -1e30f;
    for (int j = 0; j < EE; ++j) {
      float e0 = sdh[head][i] + ssh[head][j];
      e0 = e0 >= 0.f ? e0 : 0.2f * e0;
      ev[j] = e0;
      mx = fmaxf(mx, e0);
    }
    float den = 0.f;
    for (int j = 0; j < EE; ++j) { ev[j] = __expf(ev[j] - mx); den += ev[j]; }
    float inv = 1.f / den;
    for (int j = 0; j < EE; ++j) alph[head][i][j] = ev[j] * inv;
  }
  __syncthreads();

  const int row = tid >> 3, sub = tid & 7;
  float vals[32], ps = 0.f, pq = 0.f;
  for (int q = 0; q < 32; ++q) {
    int c = sub * 32 + q, head = c >> 7, o = c & 127;
    float s = 0.f;
    for (int j = 0; j < EE; ++j) s += alph[head][row][j] * hsh[head][j][o];
    s += b1[c];
    vals[q] = s; ps += s; pq += s * s;
  }
  rsum[row][sub] = ps; rsq[row][sub] = pq;
  __syncthreads();
  float mu = 0.f, sq = 0.f;
  for (int t = 0; t < 8; ++t) { mu += rsum[row][t]; sq += rsq[row][t]; }
  mu *= (1.f / 256.f);
  float var = sq * (1.f / 256.f) - mu * mu;
  float inv = rsqrtf(var + 1e-5f);
  for (int q = 0; q < 32; ++q) {
    int c = sub * 32 + q;
    float y = (vals[q] - mu) * inv * g1[c] + be1[c];
    y = y > 0.f ? y : (__expf(y) - 1.f);
    h1out[((size_t)b * EE + row) * 256 + c] = y;
  }
}

// -----------------------------------------------------------------------------
// Kernel 5: GAT layer 2 (1 head, mean) + LN + ELU -> xg (B,E,64)
// -----------------------------------------------------------------------------
__global__ __launch_bounds__(256) void gat2_kernel(
    const float* __restrict__ h1, const float* __restrict__ W2,
    const float* __restrict__ a_src2, const float* __restrict__ a_dst2,
    const float* __restrict__ b2, const float* __restrict__ g2,
    const float* __restrict__ be2, float* __restrict__ xg)
{
  const int b = blockIdx.x;
  const int tid = threadIdx.x;

  __shared__ float hsh[EE][64];
  __shared__ float ssh[EE], sdh[EE];
  __shared__ float alph[EE][EE];
  __shared__ float rsum[EE][8], rsq[EE][8];

  for (int i = tid; i < EE * 64; i += 256) {
    int e = i >> 6, o = i & 63;
    const float* hp = h1 + ((size_t)b * EE + e) * 256;
    float s = 0.f;
    for (int f = 0; f < 256; ++f) s += hp[f] * W2[f * 64 + o];
    hsh[e][o] = s;
  }
  __syncthreads();

  if (tid < EE) {
    float s1 = 0.f, s2 = 0.f;
    for (int o = 0; o < 64; ++o) {
      float hv = hsh[tid][o];
      s1 += hv * a_src2[o];
      s2 += hv * a_dst2[o];
    }
    ssh[tid] = s1; sdh[tid] = s2;
  }
  __syncthreads();

  if (tid < EE) {
    float ev[EE], mx = -1e30f;
    for (int j = 0; j < EE; ++j) {
      float e0 = sdh[tid] + ssh[j];
      e0 = e0 >= 0.f ? e0 : 0.2f * e0;
      ev[j] = e0; mx = fmaxf(mx, e0);
    }
    float den = 0.f;
    for (int j = 0; j < EE; ++j) { ev[j] = __expf(ev[j] - mx); den += ev[j]; }
    float inv = 1.f / den;
    for (int j = 0; j < EE; ++j) alph[tid][j] = ev[j] * inv;
  }
  __syncthreads();

  const int row = tid >> 3, sub = tid & 7;
  float vals[8], ps = 0.f, pq = 0.f;
  for (int q = 0; q < 8; ++q) {
    int c = sub * 8 + q;
    float s = 0.f;
    for (int j = 0; j < EE; ++j) s += alph[row][j] * hsh[j][c];
    s += b2[c]; vals[q] = s; ps += s; pq += s * s;
  }
  rsum[row][sub] = ps; rsq[row][sub] = pq;
  __syncthreads();
  float mu = 0.f, sq = 0.f;
  for (int t = 0; t < 8; ++t) { mu += rsum[row][t]; sq += rsq[row][t]; }
  mu *= (1.f / 64.f);
  float var = sq * (1.f / 64.f) - mu * mu;
  float inv = rsqrtf(var + 1e-5f);
  for (int q = 0; q < 8; ++q) {
    int c = sub * 8 + q;
    float y = (vals[q] - mu) * inv * g2[c] + be2[c];
    y = y > 0.f ? y : (__expf(y) - 1.f);
    xg[((size_t)b * EE + row) * 64 + c] = y;
  }
}

// -----------------------------------------------------------------------------
// Kernel 6: relation scores. One block per batch item (128 pairs, 8 waves).
// -----------------------------------------------------------------------------
__global__ __launch_bounds__(256) void rel_kernel(
    const float* __restrict__ xg, const int* __restrict__ pi,
    const int* __restrict__ pj, const int* __restrict__ plab,
    const float* __restrict__ rel_emb, const float* __restrict__ cc,
    const float* __restrict__ br1, const float* __restrict__ Wr1,
    const float* __restrict__ Wr2, const float* __restrict__ br2,
    float* __restrict__ scores)
{
  const int b = blockIdx.x;
  const int tid = threadIdx.x;

  __shared__ __align__(32) __bf16 As[PP][160];        // 40 KB
  __shared__ __align__(32) __bf16 BswR[8 * 32 * 16];  // 8 KB: [nt][lane][16]
  __shared__ int i1s[PP], i2s[PP], labs[PP];
  __shared__ float sred[PP];

  if (tid < PP) {
    int lab = plab[b * PP + tid];
    int a = pi[b * PP + tid], c = pj[b * PP + tid];
    int swap = (lab == FOUNDED_BY);
    i1s[tid] = swap ? c : a;
    i2s[tid] = swap ? a : c;
    labs[tid] = lab;
    sred[tid] = 0.f;
  }
  __syncthreads();

  for (int idx = tid; idx < PP * 160; idx += 256) {
    int p = idx / 160, c = idx % 160;
    float v;
    if (c < 64)       v = xg[((size_t)b * EE + i1s[p]) * 64 + c];
    else if (c < 128) v = xg[((size_t)b * EE + i2s[p]) * 64 + (c - 64)];
    else              v = rel_emb[labs[p] * 32 + (c - 128)];
    As[p][c] = (__bf16)v;
  }
  __syncthreads();

  const int wave = tid >> 5, lane = tid & 31;
  const int mt  = wave;
  const int lhi = lane >> 4, llo = lane & 15;

  for (int half = 0; half < 2; ++half) {
    v8f acc[8];
    for (int a = 0; a < 8; ++a)
      for (int k = 0; k < 8; ++k) acc[a][k] = 0.f;

    for (int k0 = 0; k0 < 160; k0 += 32) {
      for (int g = tid; g < 8 * 128; g += 256) {
        int n = g & 127, k = (g >> 7) * 4;
        const float* src = &Wr1[(size_t)(k0 + k) * 256 + half * 128 + n];
        v4bf v;
        v[0] = (__bf16)src[0];   v[1] = (__bf16)src[256];
        v[2] = (__bf16)src[512]; v[3] = (__bf16)src[768];
        int lh = (k >> 3) & 1, i0 = (k & 7) + ((k >> 4) & 1) * 8;
        int ntile = n >> 4, ll = n & 15;
        *(v4bf*)&BswR[(((ntile * 32) + lh * 16 + ll) << 4) + i0] = v;
      }
      __syncthreads();

      v8bf alo = *(const v8bf*)&As[mt * 16 + llo][k0 + lhi * 8];
      v8bf ahi = *(const v8bf*)&As[mt * 16 + llo][k0 + lhi * 8 + 16];
      v16bf af = cat16(alo, ahi);
      const __bf16* bbase = &BswR[lane << 4];
      v16bf bfr[8];
      #pragma unroll
      for (int nt = 0; nt < 8; ++nt)
        bfr[nt] = *(const v16bf*)(bbase + nt * 512);  // burst of ds_load_b128
      #pragma unroll
      for (int nt = 0; nt < 8; ++nt)                  // back-to-back WMMAs
        acc[nt] = __builtin_amdgcn_wmma_f32_16x16x32_bf16(
            false, af, false, bfr[nt], (short)0, acc[nt], false, false);
      __syncthreads();
    }

    for (int nt = 0; nt < 8; ++nt) {
      int col = half * 128 + nt * 16 + llo;
      float w2 = Wr2[col];
      float cb = cc[b * 256 + col] + br1[col];
      for (int r = 0; r < 8; ++r) {
        int row = mt * 16 + r + 8 * lhi;
        float t = acc[nt][r] + cb;
        t = t > 0.f ? t : 0.f;
        atomicAdd(&sred[row], t * w2);   // ds_add_f32
      }
    }
  }
  __syncthreads();
  if (tid < PP) scores[b * PP + tid] = sred[tid] + br2[0];
}

// -----------------------------------------------------------------------------
extern "C" void kernel_launch(void* const* d_in, const int* in_sizes, int n_in,
                              void* d_out, int out_size, void* d_ws, size_t ws_size,
                              hipStream_t stream)
{
  (void)in_sizes; (void)n_in; (void)out_size; (void)ws_size;

  const float* seq     = (const float*)d_in[0];
  const int*   starts  = (const int*)d_in[1];
  const int*   lens    = (const int*)d_in[2];
  const int*   types   = (const int*)d_in[3];
  const int*   pi      = (const int*)d_in[4];
  const int*   pj      = (const int*)d_in[5];
  const int*   plab    = (const int*)d_in[6];
  const float* type_emb= (const float*)d_in[7];
  const float* rel_emb = (const float*)d_in[8];
  const float* W1      = (const float*)d_in[9];
  const float* a_src1  = (const float*)d_in[10];
  const float* a_dst1  = (const float*)d_in[11];
  const float* b1      = (const float*)d_in[12];
  const float* g1      = (const float*)d_in[13];
  const float* be1     = (const float*)d_in[14];
  const float* W2      = (const float*)d_in[15];
  const float* a_src2  = (const float*)d_in[16];
  const float* a_dst2  = (const float*)d_in[17];
  const float* b2      = (const float*)d_in[18];
  const float* g2      = (const float*)d_in[19];
  const float* be2     = (const float*)d_in[20];
  const float* Wn1     = (const float*)d_in[21];
  const float* bn1     = (const float*)d_in[22];
  const float* Wn2     = (const float*)d_in[23];
  const float* bn2     = (const float*)d_in[24];
  const float* Wr1     = (const float*)d_in[25];
  const float* br1     = (const float*)d_in[26];
  const float* Wr2     = (const float*)d_in[27];
  const float* br2     = (const float*)d_in[28];

  float* out    = (float*)d_out;
  float* ner    = out;                         // (B,S,NNER)
  float* scores = out + (size_t)BB * SS * NNER;

  float* ws = (float*)d_ws;
  float* x  = ws;                              // (B,E,H)
  float* h1 = x  + (size_t)BB * EE * HH;       // (B,E,256)
  float* xg = h1 + (size_t)BB * EE * 256;      // (B,E,64)
  float* cc = xg + (size_t)BB * EE * 64;       // (B,256)

  ner_kernel<<<dim3((BB * SS) / 32), dim3(128), 0, stream>>>(seq, Wn1, bn1, Wn2, bn2, ner);
  span_pool_kernel<<<dim3(BB * EE), dim3(256), 0, stream>>>(seq, starts, lens, types, type_emb, x);
  cls_kernel<<<dim3(BB), dim3(256), 0, stream>>>(seq, Wr1, cc);
  gat1_kernel<<<dim3(BB), dim3(256), 0, stream>>>(x, W1, a_src1, a_dst1, b1, g1, be1, h1);
  gat2_kernel<<<dim3(BB), dim3(256), 0, stream>>>(h1, W2, a_src2, a_dst2, b2, g2, be2, xg);
  rel_kernel<<<dim3(BB), dim3(256), 0, stream>>>(xg, pi, pj, plab, rel_emb, cc, br1, Wr1, Wr2, br2, scores);
}
